// SpeakerDecoderTran_40827959116513
// MI455X (gfx1250) — compile-verified, matrix operand order checked by reading, not run on hardware
//
#include <hip/hip_runtime.h>
#include <hip/hip_bf16.h>

typedef __bf16 bf16;
typedef __attribute__((ext_vector_type(16))) __bf16 v16bf;
typedef __attribute__((ext_vector_type(8)))  __bf16 v8bf;
typedef __attribute__((ext_vector_type(8)))  float   v8f;

#define B_  16
#define T_  128
#define E_  512
#define H_  1024
#define V_  32000
#define CL_ 49
#define CD_ 2048
#define M_  (B_*T_)   /* 2048 rows of the sequence-flattened activations */

__device__ __forceinline__ v16bf cat16(v8bf lo, v8bf hi) {
  return __builtin_shufflevector(lo, hi, 0,1,2,3,4,5,6,7,8,9,10,11,12,13,14,15);
}
__device__ __forceinline__ float sigmoidf(float x) { return 1.0f / (1.0f + __expf(-x)); }
__device__ __forceinline__ unsigned lds_lo32(const void* p) {
  // low 32 bits of a flat shared-memory address == wave-relative LDS byte address
  return (unsigned)(unsigned long long)p;
}

// ---------------------------------------------------------------------------
// fp32 -> bf16 conversion (weights, ctx, h0)
// ---------------------------------------------------------------------------
__global__ __launch_bounds__(256) void k_cvt_bf16(const float* __restrict__ src,
                                                  bf16* __restrict__ dst, long n) {
  long i = (long)blockIdx.x * 256 + threadIdx.x;
  if (i < n) dst[i] = (bf16)src[i];
}

// ---------------------------------------------------------------------------
// Embedding gather: words[B,T] -> xemb_bf16 [2048, 512]
// ---------------------------------------------------------------------------
__global__ __launch_bounds__(256) void k_embed(const int* __restrict__ words,
                                               const float* __restrict__ emb,
                                               bf16* __restrict__ xemb) {
  long i = (long)blockIdx.x * 256 + threadIdx.x;   // over 2048*512, exact grid
  int m = (int)(i >> 9);
  int e = (int)(i & 511);
  int w = words[m];
  xemb[i] = (bf16)emb[(long)w * E_ + e];
}

// ---------------------------------------------------------------------------
// Generic WMMA GEMM:  C[M,N] = act( A[M,K](bf16) * B[N,K]^T(bf16) + bias0 + bias1 )
// Block = 128 threads (4 waves).  Block tile 64(M) x 64(N); wave w owns rows
// m0+16w and all 64 N columns as four 16x16 accumulators.
// B tile (64x32 bf16, shared by all waves) is staged to LDS with
// global_load_async_to_lds_b128, double buffered on ASYNCcnt (async loads
// complete in order, so asynccnt<=2 releases the current buffer while the
// next-tile loads are still in flight).
// ---------------------------------------------------------------------------
__device__ __forceinline__ void stage_b_async(const bf16* __restrict__ Bm, int ldb,
                                              int N, int n0, int k0,
                                              bf16* dstLds, int tid) {
  // 64 rows x 32 k = 4KB = 256 chunks of 16B ; 128 threads -> 2 chunks each
#pragma unroll
  for (int it = 0; it < 2; ++it) {
    int c   = tid + it * 128;
    int row = n0 + (c >> 2);
    if (row > N - 1) row = N - 1;
    int ko  = (c & 3) * 8;
    const bf16* gp = Bm + (long)row * ldb + k0 + ko;
    unsigned la = lds_lo32(dstLds + c * 8);
    asm volatile("global_load_async_to_lds_b128 %0, %1, off"
                 :: "v"(la), "v"(gp) : "memory");
  }
}

// Load A fragment + all 4 B fragments first (one dscnt wait covers them all),
// then issue the 4 WMMAs back-to-back so the XDL pipe stays fed.
__device__ __forceinline__ void gemm_compute_step(const bf16* __restrict__ ap, int kk,
                                                  const bf16* __restrict__ shTile,
                                                  int lrow, int half, v8f acc[4]) {
  // ISA 16-bit A layout: lanes 0-15: K=kk+0..7 , kk+16..23 ;
  //                      lanes 16-31: K=kk+8..15 , kk+24..31
  v8bf alo = *(const v8bf*)(ap + kk + half * 8);
  v8bf ahi = *(const v8bf*)(ap + kk + 16 + half * 8);
  v16bf afrag = cat16(alo, ahi);
  v16bf bfrag[4];
#pragma unroll
  for (int j = 0; j < 4; ++j) {
    const bf16* bp = shTile + (j * 16 + lrow) * 32;
    bfrag[j] = cat16(*(const v8bf*)(bp + half * 8),
                     *(const v8bf*)(bp + 16 + half * 8));
  }
#pragma unroll
  for (int j = 0; j < 4; ++j)
    acc[j] = __builtin_amdgcn_wmma_f32_16x16x32_bf16(
                 false, afrag, false, bfrag[j], (short)0, acc[j], false, false);
}

__global__ __launch_bounds__(128)
void k_gemm(const bf16* __restrict__ A, int lda,
            const bf16* __restrict__ Bm, int ldb,
            int M, int N, int K,
            const float* __restrict__ bias0, const float* __restrict__ bias1,
            int act, float* __restrict__ Cf, bf16* __restrict__ Cbf, int ldc) {
  __shared__ bf16 shB[2][64 * 32];   // 8 KB double buffer

  const int tid  = threadIdx.x;
  const int wave = tid >> 5;
  const int lane = tid & 31;
  const int lrow = lane & 15;
  const int half = lane >> 4;
  const int n0   = blockIdx.x * 64;
  const int m0   = blockIdx.y * 64 + wave * 16;

  int arow = m0 + lrow;                  // clamp for ragged M (M=784 case)
  if (arow > M - 1) arow = M - 1;
  const bf16* ap = A + (long)arow * lda;

  v8f acc[4] = {v8f{}, v8f{}, v8f{}, v8f{}};

  stage_b_async(Bm, ldb, N, n0, 0, &shB[0][0], tid);

  const int ksteps = K >> 5;
  int ks = 0;
  // steady state: branchless body, always prefetching the next B tile
  for (; ks < ksteps - 1; ++ks) {
    const int kk  = ks << 5;
    const int cur = ks & 1;
    stage_b_async(Bm, ldb, N, n0, kk + 32, &shB[cur ^ 1][0], tid);
    asm volatile("s_wait_asynccnt 2" ::: "memory");   // current tile landed
    __syncthreads();
    __builtin_prefetch(ap + kk + 64, 0, 0);           // A stream -> L2
    gemm_compute_step(ap, kk, &shB[cur][0], lrow, half, acc);
    __syncthreads();
  }
  // tail iteration
  {
    const int kk  = ks << 5;
    const int cur = ks & 1;
    asm volatile("s_wait_asynccnt 0" ::: "memory");
    __syncthreads();
    gemm_compute_step(ap, kk, &shB[cur][0], lrow, half, acc);
  }

#pragma unroll
  for (int j = 0; j < 4; ++j) {
    int col = n0 + j * 16 + lrow;        // N is always a multiple of 64 here
    float bsum = (bias0 ? bias0[col] : 0.0f) + (bias1 ? bias1[col] : 0.0f);
#pragma unroll
    for (int g = 0; g < 8; ++g) {
      int row = m0 + g + half * 8;       // C/D layout: VGPR g <-> M=g / M=8+g
      if (row < M) {
        float v = acc[j][g] + bsum;
        if (act == 1) v = tanhf(v);
        if (Cf)  Cf[(long)row * ldc + col]  = v;
        if (Cbf) Cbf[(long)row * ldc + col] = (bf16)v;
      }
    }
  }
}

// ---------------------------------------------------------------------------
// One LSTM time step.  64 blocks x 1 wave; block owns hidden columns
// jc0..jc0+15 and computes all four gate tiles (i,f,g,o) with WMMA:
//   gates[16, 16] += h_bf[16,1024] @ W_hh[g*1024+jc, 1024]^T
// then the elementwise cell update.  h is double-buffered across launches.
// ---------------------------------------------------------------------------
__global__ __launch_bounds__(32)
void k_lstm_step(const bf16* __restrict__ hbf_in,   // [16,1024]
                 const bf16* __restrict__ Whh,      // [4096,1024] bf16
                 const float* __restrict__ xg,      // [2048,4096] x-contrib + biases
                 float* __restrict__ c_st,          // [16,1024] cell state (in/out)
                 float* __restrict__ h_st,          // [16,1024] fp32 h (for h1 output)
                 bf16* __restrict__ hbf_out,        // [16,1024]
                 bf16* __restrict__ cat_bf,         // [2048,2048] ([att | x])
                 int t) {
  const int lane = threadIdx.x & 31;
  const int lrow = lane & 15;
  const int half = lane >> 4;
  const int jc0  = blockIdx.x * 16;

  v8f acc[4] = {v8f{}, v8f{}, v8f{}, v8f{}};
  const bf16* ap = hbf_in + lrow * H_;

  for (int kk = 0; kk < H_; kk += 32) {
    v8bf alo = *(const v8bf*)(ap + kk + half * 8);
    v8bf ahi = *(const v8bf*)(ap + kk + 16 + half * 8);
    v16bf afrag = cat16(alo, ahi);
    v16bf bfrag[4];
#pragma unroll
    for (int g = 0; g < 4; ++g) {
      const bf16* bp = Whh + (long)(g * H_ + jc0 + lrow) * H_ + kk;
      bfrag[g] = cat16(*(const v8bf*)(bp + half * 8),
                       *(const v8bf*)(bp + 16 + half * 8));
    }
#pragma unroll
    for (int g = 0; g < 4; ++g)
      acc[g] = __builtin_amdgcn_wmma_f32_16x16x32_bf16(
                   false, afrag, false, bfrag[g], (short)0, acc[g], false, false);
  }

#pragma unroll
  for (int g8 = 0; g8 < 8; ++g8) {
    int r    = g8 + half * 8;            // batch row 0..15
    int cidx = jc0 + lrow;               // hidden column
    long xrow = ((long)r * T_ + t) * (4 * H_);
    float gi = acc[0][g8] + xg[xrow + 0 * H_ + cidx];
    float gf = acc[1][g8] + xg[xrow + 1 * H_ + cidx];
    float gg = acc[2][g8] + xg[xrow + 2 * H_ + cidx];
    float go = acc[3][g8] + xg[xrow + 3 * H_ + cidx];
    float ii = sigmoidf(gi), ff = sigmoidf(gf);
    float cc = tanhf(gg),   oo = sigmoidf(go);
    int hi = r * H_ + cidx;
    float cnew = ff * c_st[hi] + ii * cc;
    float hnew = oo * tanhf(cnew);
    c_st[hi] = cnew;
    h_st[hi] = hnew;
    hbf_out[hi] = (bf16)hnew;
    cat_bf[((long)r * T_ + t) * (2 * H_) + H_ + cidx] = (bf16)hnew;  // x half
  }
}

// ---------------------------------------------------------------------------
// Attention (L=49, <0.5% of FLOPs -> plain VALU).  One block per (b,t).
// scores -> softmax -> att ; att written as bf16 into cat[:, 0:H].
// ---------------------------------------------------------------------------
__global__ __launch_bounds__(64)
void k_attn(const float* __restrict__ q,   // [2048,1024]
            const float* __restrict__ km,  // [784,1024]
            const float* __restrict__ vm,  // [784,1024]
            bf16* __restrict__ cat_bf) {   // [2048,2048]
  const int m = blockIdx.x;                // b*T + t
  const int b = m >> 7;
  const int tid = threadIdx.x;
  __shared__ float sp[64];
  __shared__ float smax, sinv;

  if (tid < CL_) {
    const float* qp = q + (long)m * H_;
    const float* kp = km + ((long)b * CL_ + tid) * H_;
    float s = 0.0f;
    for (int i = 0; i < H_; ++i) s += qp[i] * kp[i];
    sp[tid] = s * 0.03125f;                // 1/sqrt(H)=1/32
  }
  __syncthreads();
  if (tid == 0) {
    float mx = sp[0];
    for (int l = 1; l < CL_; ++l) mx = fmaxf(mx, sp[l]);
    smax = mx;
  }
  __syncthreads();
  if (tid < CL_) sp[tid] = __expf(sp[tid] - smax);
  __syncthreads();
  if (tid == 0) {
    float s = 0.0f;
    for (int l = 0; l < CL_; ++l) s += sp[l];
    sinv = 1.0f / s;
  }
  __syncthreads();
  if (tid < CL_) sp[tid] *= sinv;
  __syncthreads();

  for (int c = tid; c < H_; c += 64) {
    float a = 0.0f;
    for (int l = 0; l < CL_; ++l)
      a += sp[l] * vm[((long)b * CL_ + l) * H_ + c];
    cat_bf[(long)m * (2 * H_) + c] = (bf16)a;   // att half of concat
  }
}

// ---------------------------------------------------------------------------
extern "C" void kernel_launch(void* const* d_in, const int* in_sizes, int n_in,
                              void* d_out, int out_size, void* d_ws, size_t ws_size,
                              hipStream_t stream) {
  (void)in_sizes; (void)n_in; (void)out_size; (void)ws_size;

  const int*   words = (const int*)  d_in[0];
  const float* h0    = (const float*)d_in[1];
  const float* c0    = (const float*)d_in[2];
  const float* ctx   = (const float*)d_in[3];
  const float* emb   = (const float*)d_in[4];
  const float* W_ih  = (const float*)d_in[5];
  const float* W_hh  = (const float*)d_in[6];
  const float* b_ih  = (const float*)d_in[7];
  const float* b_hh  = (const float*)d_in[8];
  const float* Wq    = (const float*)d_in[9];
  const float* bq    = (const float*)d_in[10];
  const float* Wk    = (const float*)d_in[11];
  const float* bk    = (const float*)d_in[12];
  const float* Wv    = (const float*)d_in[13];
  const float* bv    = (const float*)d_in[14];
  const float* Wo    = (const float*)d_in[15];
  const float* bo    = (const float*)d_in[16];
  const float* Wp    = (const float*)d_in[17];
  const float* bp    = (const float*)d_in[18];
  float* out = (float*)d_out;

  // ---- workspace carve (256B aligned) ----
  size_t off = 0;
  char* base = (char*)d_ws;
  auto carve = [&](size_t bytes) -> void* {
    off = (off + 255) & ~(size_t)255;
    void* r = base + off;
    off += bytes;
    return r;
  };
  bf16* Wih_bf  = (bf16*)carve((size_t)4*H_*E_      * 2);
  bf16* Whh_bf  = (bf16*)carve((size_t)4*H_*H_      * 2);
  bf16* Wq_bf   = (bf16*)carve((size_t)H_*H_        * 2);
  bf16* Wk_bf   = (bf16*)carve((size_t)H_*CD_       * 2);
  bf16* Wv_bf   = (bf16*)carve((size_t)H_*CD_       * 2);
  bf16* Wo_bf   = (bf16*)carve((size_t)H_*2*H_      * 2);
  bf16* Wp_bf   = (bf16*)carve((size_t)V_*H_        * 2);
  bf16* ctx_bf  = (bf16*)carve((size_t)B_*CL_*CD_   * 2);
  bf16* xemb_bf = (bf16*)carve((size_t)M_*E_        * 2);
  float* xg     = (float*)carve((size_t)M_*4*H_     * 4);
  float* c_st   = (float*)carve((size_t)B_*H_       * 4);
  float* h_st   = (float*)carve((size_t)B_*H_       * 4);
  bf16* hbuf0   = (bf16*)carve((size_t)B_*H_        * 2);
  bf16* hbuf1   = (bf16*)carve((size_t)B_*H_        * 2);
  bf16* cat_bf  = (bf16*)carve((size_t)M_*2*H_      * 2);
  float* q_f    = (float*)carve((size_t)M_*H_       * 4);
  float* k_f    = (float*)carve((size_t)B_*CL_*H_   * 4);
  float* v_f    = (float*)carve((size_t)B_*CL_*H_   * 4);
  bf16* mrg_bf  = (bf16*)carve((size_t)M_*H_        * 2);

  auto cvt = [&](const float* s, bf16* d, long n) {
    k_cvt_bf16<<<(unsigned)((n + 255) / 256), 256, 0, stream>>>(s, d, n);
  };

  // ---- one-time bf16 conversions ----
  cvt(W_ih, Wih_bf, (long)4*H_*E_);
  cvt(W_hh, Whh_bf, (long)4*H_*H_);
  cvt(Wq,   Wq_bf,  (long)H_*H_);
  cvt(Wk,   Wk_bf,  (long)H_*CD_);
  cvt(Wv,   Wv_bf,  (long)H_*CD_);
  cvt(Wo,   Wo_bf,  (long)H_*2*H_);
  cvt(Wp,   Wp_bf,  (long)V_*H_);
  cvt(ctx,  ctx_bf, (long)B_*CL_*CD_);
  cvt(h0,   hbuf0,  (long)B_*H_);

  // ---- embedding gather ----
  k_embed<<<(M_*E_)/256, 256, 0, stream>>>(words, emb, xemb_bf);

  // ---- xg = x_emb @ W_ih^T + (b_ih + b_hh)  [2048,4096] ----
  k_gemm<<<dim3(4*H_/64, M_/64), 128, 0, stream>>>(
      xemb_bf, E_, Wih_bf, E_, M_, 4*H_, E_, b_ih, b_hh, 0, xg, nullptr, 4*H_);

  // ---- init states ----
  hipMemcpyAsync(c_st, c0, (size_t)B_*H_*4, hipMemcpyDeviceToDevice, stream);

  // ---- LSTM recurrence, 128 sequential steps ----
  for (int t = 0; t < T_; ++t) {
    bf16* hin  = (t & 1) ? hbuf1 : hbuf0;
    bf16* hout = (t & 1) ? hbuf0 : hbuf1;
    k_lstm_step<<<H_/16, 32, 0, stream>>>(hin, Whh_bf, xg, c_st, h_st, hout,
                                          cat_bf, t);
  }

  // ---- q / k / v projections ----
  k_gemm<<<dim3(H_/64, M_/64), 128, 0, stream>>>(
      cat_bf + H_, 2*H_, Wq_bf, H_, M_, H_, H_, bq, nullptr, 0, q_f, nullptr, H_);
  k_gemm<<<dim3(H_/64, (B_*CL_ + 63)/64), 128, 0, stream>>>(
      ctx_bf, CD_, Wk_bf, CD_, B_*CL_, H_, CD_, bk, nullptr, 0, k_f, nullptr, H_);
  k_gemm<<<dim3(H_/64, (B_*CL_ + 63)/64), 128, 0, stream>>>(
      ctx_bf, CD_, Wv_bf, CD_, B_*CL_, H_, CD_, bv, nullptr, 0, v_f, nullptr, H_);

  // ---- softmax attention, writes att into cat[:, 0:H] ----
  k_attn<<<M_, 64, 0, stream>>>(q_f, k_f, v_f, cat_bf);

  // ---- merged = tanh(cat @ Wo^T + bo) -> bf16 ----
  k_gemm<<<dim3(H_/64, M_/64), 128, 0, stream>>>(
      cat_bf, 2*H_, Wo_bf, 2*H_, M_, H_, 2*H_, bo, nullptr, 1, nullptr, mrg_bf, H_);

  // ---- logit = merged @ Wp^T + bp -> d_out fp32 [2048, 32000] ----
  k_gemm<<<dim3(V_/64, M_/64), 128, 0, stream>>>(
      mrg_bf, H_, Wp_bf, H_, M_, V_, H_, bp, nullptr, 0, out, nullptr, V_);

  // ---- h1, c1 tail outputs ----
  hipMemcpyAsync(out + (size_t)M_*V_,            h_st, (size_t)B_*H_*4,
                 hipMemcpyDeviceToDevice, stream);
  hipMemcpyAsync(out + (size_t)M_*V_ + B_*H_,    c_st, (size_t)B_*H_*4,
                 hipMemcpyDeviceToDevice, stream);
}
